// RefineAttention_14826227105920
// MI455X (gfx1250) — matrix operand order repaired and checked by reading, not compiled, not run on hardware
//
#include <hip/hip_runtime.h>
#include <hip/hip_bf16.h>

// CDNA5 / gfx1250: wave32, WMMA 16x16x32 bf16 with f32 accumulate.
typedef __attribute__((ext_vector_type(16))) __bf16        v16bf;
typedef __attribute__((ext_vector_type(8)))  float         v8f;
typedef __attribute__((ext_vector_type(4)))  unsigned int  v4u;
typedef __attribute__((ext_vector_type(4)))  float         v4f;

union BFrag { unsigned short us[16]; v4u q[2]; v16bf v; };

__device__ __forceinline__ unsigned short f2bf(float f) {
    unsigned int u = __float_as_uint(f);
    u += 0x7FFFu + ((u >> 16) & 1u);          // round-to-nearest-even
    return (unsigned short)(u >> 16);
}

__device__ __forceinline__ v4u packbf8(const v8f& s) {   // 8 f32 -> 8 bf16 in 16B
    v4u w;
#pragma unroll
    for (int i = 0; i < 4; ++i)
        w[i] = (unsigned)f2bf(s[2 * i]) | ((unsigned)f2bf(s[2 * i + 1]) << 16);
    return w;
}

#define LOG2E 1.44269504088896340736f
#define QSCALE 0.17677669529663687f           /* 1/sqrt(hd=32) */

// ---------------------------------------------------------------------------
// Kernel 1: QKV projection. qkv[d, tok] = sum_c x[tok,c]*w[d,c]
// d = h*96 + t (t<32:Q, <64:K, else V). Q/K stored [B,H,N,32] bf16 (scale
// folded into Q); V stored TRANSPOSED [B,H,32,N] so the attention kernel can
// stage its V tile with b128 copies.
// ---------------------------------------------------------------------------
__global__ __launch_bounds__(128) void qkv_gemm_bf16(
    const float* __restrict__ x, const float* __restrict__ w,
    unsigned short* __restrict__ Q, unsigned short* __restrict__ K,
    unsigned short* __restrict__ V)
{
    __shared__ __align__(16) unsigned short Xl[16 * 264];
    __shared__ __align__(16) unsigned short Wl[64 * 264];
    const int tid = threadIdx.x, lane = tid & 31, wv = tid >> 5;
    const int mt = blockIdx.x;   // token tile (16 tokens)
    const int nt = blockIdx.y;   // 64-wide column tile of 768

    const float2* x2 = (const float2*)x;
    const float2* w2 = (const float2*)w;
    for (int e = tid; e < 16 * 128; e += 128) {
        int r = e >> 7, cp = e & 127;
        float2 f = x2[(size_t)(mt * 16 + r) * 128 + cp];
        *(unsigned int*)&Xl[r * 264 + 2 * cp] =
            (unsigned)f2bf(f.x) | ((unsigned)f2bf(f.y) << 16);
    }
    for (int e = tid; e < 64 * 128; e += 128) {
        int r = e >> 7, cp = e & 127;
        float2 f = w2[(size_t)(nt * 64 + r) * 128 + cp];
        *(unsigned int*)&Wl[r * 264 + 2 * cp] =
            (unsigned)f2bf(f.x) | ((unsigned)f2bf(f.y) << 16);
    }
    __syncthreads();

    const int jl = lane & 15, kh = lane >> 4;
    v8f acc = {0.f, 0.f, 0.f, 0.f, 0.f, 0.f, 0.f, 0.f};
    for (int kc = 0; kc < 8; ++kc) {
        BFrag a, b;
        const v4u* pa = (const v4u*)&Xl[jl * 264 + kc * 32 + kh * 8];
        a.q[0] = pa[0]; a.q[1] = pa[2];                       // K runs 0..7, 16..23
        const v4u* pb = (const v4u*)&Wl[(wv * 16 + jl) * 264 + kc * 32 + kh * 8];
        b.q[0] = pb[0]; b.q[1] = pb[2];                       // K runs 0..7, 16..23
        acc = __builtin_amdgcn_wmma_f32_16x16x32_bf16(
            false, a.v, false, b.v, (short)0, acc, false, false);
    }

    const int d = nt * 64 + wv * 16 + jl;
    const int h = d / 96, t = d % 96;
#pragma unroll
    for (int r = 0; r < 8; ++r) {
        int tok = mt * 16 + r + 8 * kh;        // C/D layout: row = r + 8*(lane>>4)
        int bb = tok >> 11, n = tok & 2047;
        float vv = acc[r];
        if (t < 32)
            Q[((size_t)(bb * 8 + h) * 2048 + n) * 32 + t] = f2bf(vv * QSCALE);
        else if (t < 64)
            K[((size_t)(bb * 8 + h) * 2048 + n) * 32 + (t - 32)] = f2bf(vv);
        else
            V[((size_t)(bb * 8 + h) * 32 + (t - 64)) * 2048 + n] = f2bf(vv);
    }
}

// ---------------------------------------------------------------------------
// Kernel 2: fused flash attention (softmax over keys). 128 thr = 4 waves,
// 64 queries/block, 32-key chunks. All fragment traffic is ds_load/store_b128.
// ---------------------------------------------------------------------------
__global__ __launch_bounds__(128) void attn_flash(
    const unsigned short* __restrict__ Q, const unsigned short* __restrict__ K,
    const unsigned short* __restrict__ V, unsigned short* __restrict__ AO)
{
    __shared__ __align__(16) unsigned short Ql[64 * 40];
    __shared__ __align__(16) unsigned short Kl[32 * 40];
    __shared__ __align__(16) unsigned short Vt[32 * 40];   // [d][i]
    __shared__ __align__(16) unsigned short Pl[4][16 * 40]; // per wave, [j][i]

    const int tid = threadIdx.x, lane = tid & 31, wv = tid >> 5;
    const int bh = blockIdx.x;            // 0..31 = b*8+h
    const int qb = blockIdx.y * 64;       // query block base
    const unsigned short* Qp = Q + (size_t)bh * 2048 * 32 + (size_t)qb * 32;
    const unsigned short* Kp = K + (size_t)bh * 2048 * 32;
    const unsigned short* Vp = V + (size_t)bh * 32 * 2048; // [32][2048]

    { // stage 64x32 query tile -> padded LDS (b128 copies)
        const v4u* s = (const v4u*)Qp;
        for (int u = tid; u < 256; u += 128)
            ((v4u*)(Ql + (u >> 2) * 40))[u & 3] = s[u];
    }
    __syncthreads();

    const int jl = lane & 15, kh = lane >> 4;
    BFrag qf;   // B-fragment of S-gemm, loop invariant: B[d,j] = q[j][d]
    {
        const v4u* p = (const v4u*)&Ql[(wv * 16 + jl) * 40 + kh * 8];
        qf.q[0] = p[0]; qf.q[1] = p[2];
    }

    v8f o0 = {0.f, 0.f, 0.f, 0.f, 0.f, 0.f, 0.f, 0.f};
    v8f o1 = {0.f, 0.f, 0.f, 0.f, 0.f, 0.f, 0.f, 0.f};
    float mrun = -INFINITY, lrun = 0.f;

    for (int kb = 0; kb < 2048; kb += 32) {
        __syncthreads();   // previous chunk's Kl/Vt reads done
        { // stage K chunk [32 keys][32 d] (contiguous) and V chunk [32 d][32 i]
            const v4u* sk = (const v4u*)(Kp + (size_t)kb * 32);
            for (int u = tid; u < 128; u += 128)   // one b128 per thread
                ((v4u*)(Kl + (u >> 2) * 40))[u & 3] = sk[u];
            {
                int u = tid;                       // 128 units exactly
                int dr = u >> 2, qi = u & 3;
                ((v4u*)(Vt + dr * 40))[qi] =
                    ((const v4u*)(Vp + (size_t)dr * 2048 + kb))[qi];
            }
        }
        // prefetch next chunk while this one is consumed
        if (kb + 32 < 2048) {
            if (tid < 16)
                __builtin_prefetch(Kp + (size_t)(kb + 32) * 32 + tid * 128, 0, 1);
            else if (tid < 48)
                __builtin_prefetch(Vp + (size_t)(tid - 16) * 2048 + kb + 32, 0, 1);
        }
        __syncthreads();

        // ---- logits: two 16x16 key tiles, K-dim = hd = 32 ----
        v8f s0, s1;
        {
            const v8f z = {0.f, 0.f, 0.f, 0.f, 0.f, 0.f, 0.f, 0.f};
            BFrag a;
            const v4u* p0 = (const v4u*)&Kl[jl * 40 + kh * 8];
            a.q[0] = p0[0]; a.q[1] = p0[2];
            s0 = __builtin_amdgcn_wmma_f32_16x16x32_bf16(
                false, a.v, false, qf.v, (short)0, z, false, false);
            const v4u* p1 = (const v4u*)&Kl[(16 + jl) * 40 + kh * 8];
            a.q[0] = p1[0]; a.q[1] = p1[2];
            s1 = __builtin_amdgcn_wmma_f32_16x16x32_bf16(
                false, a.v, false, qf.v, (short)0, z, false, false);
        }

        // ---- online softmax over key axis (column j shared by lanes l,l^16)
        float cm = s0[0];
#pragma unroll
        for (int r = 1; r < 8; ++r) cm = fmaxf(cm, s0[r]);
#pragma unroll
        for (int r = 0; r < 8; ++r) cm = fmaxf(cm, s1[r]);
        cm = fmaxf(cm, __shfl_xor(cm, 16, 32));
        float mnew = fmaxf(mrun, cm);
        float corr = exp2f((mrun - mnew) * LOG2E);
        float ps = 0.f;
#pragma unroll
        for (int r = 0; r < 8; ++r) {
            s0[r] = exp2f((s0[r] - mnew) * LOG2E); ps += s0[r];
            s1[r] = exp2f((s1[r] - mnew) * LOG2E); ps += s1[r];
        }
        ps += __shfl_xor(ps, 16, 32);
        lrun = lrun * corr + ps;
        mrun = mnew;
#pragma unroll
        for (int r = 0; r < 8; ++r) { o0[r] *= corr; o1[r] *= corr; }

        // ---- P tile -> per-wave LDS transposed [j][i] via 2x ds_store_b128
        unsigned short* P = Pl[wv];
        *(v4u*)&P[jl * 40 + 8 * kh]      = packbf8(s0);   // i = 8*kh + 0..7
        *(v4u*)&P[jl * 40 + 16 + 8 * kh] = packbf8(s1);   // i = 16 + 8*kh + 0..7
        __syncthreads();

        BFrag pf;   // B[i,j] = P[i][j] from Pt[j][i]: two runs of 8
        {
            const v4u* p = (const v4u*)&P[jl * 40 + kh * 8];
            pf.q[0] = p[0]; pf.q[1] = p[2];
        }
        BFrag va;   // A[d,i] = Vt[d][i]: two runs of 8
        {
            const v4u* p = (const v4u*)&Vt[jl * 40 + kh * 8];
            va.q[0] = p[0]; va.q[1] = p[2];
        }
        o0 = __builtin_amdgcn_wmma_f32_16x16x32_bf16(
            false, va.v, false, pf.v, (short)0, o0, false, false);
        {
            const v4u* p = (const v4u*)&Vt[(16 + jl) * 40 + kh * 8];
            va.q[0] = p[0]; va.q[1] = p[2];
        }
        o1 = __builtin_amdgcn_wmma_f32_16x16x32_bf16(
            false, va.v, false, pf.v, (short)0, o1, false, false);
    }

    const float inv = 1.f / lrun;
    const int n = qb + wv * 16 + jl;
    const int bb = bh >> 3, h = bh & 7;
    unsigned short* dst = AO + ((size_t)bb * 2048 + n) * 256 + (size_t)h * 32;
    v8f t0, t1;
#pragma unroll
    for (int r = 0; r < 8; ++r) { t0[r] = o0[r] * inv; t1[r] = o1[r] * inv; }
    *(v4u*)&dst[8 * kh]      = packbf8(t0);   // c = h*32 + 8*kh + 0..7
    *(v4u*)&dst[16 + 8 * kh] = packbf8(t1);   // c = h*32 + 16 + 8*kh + 0..7
}

// ---------------------------------------------------------------------------
// Kernel 3: output projection + bias. out[b, dout, n] (fp32), b128 stores.
// ---------------------------------------------------------------------------
__global__ __launch_bounds__(128) void proj_gemm_bf16(
    const unsigned short* __restrict__ A, const float* __restrict__ w,
    const float* __restrict__ bias, float* __restrict__ out)
{
    __shared__ __align__(16) unsigned short Al[16 * 264];
    __shared__ __align__(16) unsigned short Wl[64 * 264];
    const int tid = threadIdx.x, lane = tid & 31, wv = tid >> 5;
    const int mt = blockIdx.x;   // token tile
    const int nt = blockIdx.y;   // 64-wide dout tile of 256

    {   // A tile already bf16, rows contiguous: b128 copies into padded LDS
        const v4u* s = (const v4u*)(A + (size_t)mt * 16 * 256);
        for (int u = tid; u < 512; u += 128)
            ((v4u*)(Al + (u >> 5) * 264))[u & 31] = s[u];
    }
    const float2* w2 = (const float2*)w;
    for (int e = tid; e < 64 * 128; e += 128) {
        int r = e >> 7, cp = e & 127;
        float2 f = w2[(size_t)(nt * 64 + r) * 128 + cp];
        *(unsigned int*)&Wl[r * 264 + 2 * cp] =
            (unsigned)f2bf(f.x) | ((unsigned)f2bf(f.y) << 16);
    }
    __syncthreads();

    const int jl = lane & 15, kh = lane >> 4;
    v8f acc = {0.f, 0.f, 0.f, 0.f, 0.f, 0.f, 0.f, 0.f};
    for (int kc = 0; kc < 8; ++kc) {
        BFrag a, b;
        const v4u* pa = (const v4u*)&Al[jl * 264 + kc * 32 + kh * 8];
        a.q[0] = pa[0]; a.q[1] = pa[2];
        const v4u* pb = (const v4u*)&Wl[(wv * 16 + jl) * 264 + kc * 32 + kh * 8];
        b.q[0] = pb[0]; b.q[1] = pb[2];
        acc = __builtin_amdgcn_wmma_f32_16x16x32_bf16(
            false, a.v, false, b.v, (short)0, acc, false, false);
    }

    const int dout = nt * 64 + wv * 16 + jl;
    const float bb = bias[dout];
    const int tok0 = mt * 16 + 8 * kh;           // rows this lane owns: tok0..+7
    const int b_ = tok0 >> 11, n0 = tok0 & 2047;
    float* dst = out + ((size_t)b_ * 256 + dout) * 2048 + n0;
    v4f f0, f1;
#pragma unroll
    for (int i = 0; i < 4; ++i) { f0[i] = acc[i] + bb; f1[i] = acc[4 + i] + bb; }
    *(v4f*)dst = f0;
    *(v4f*)(dst + 4) = f1;
}

// ---------------------------------------------------------------------------
extern "C" void kernel_launch(void* const* d_in, const int* in_sizes, int n_in,
                              void* d_out, int out_size, void* d_ws, size_t ws_size,
                              hipStream_t stream) {
    (void)in_sizes; (void)n_in; (void)out_size; (void)ws_size;
    const float* x      = (const float*)d_in[0];   // [4,2048,256]
    const float* qkv_w  = (const float*)d_in[1];   // [768,256]
    const float* proj_w = (const float*)d_in[2];   // [256,256]
    const float* proj_b = (const float*)d_in[3];   // [256]
    float* out = (float*)d_out;                    // [4,256,2048,1]

    // workspace (bf16): Q,K [B,H,N,32]; V [B,H,32,N]; AO [B,N,256] -> 16 MB
    const size_t E = (size_t)4 * 8 * 2048 * 32;    // 2,097,152 elems each
    unsigned short* Qws = (unsigned short*)d_ws;
    unsigned short* Kws = Qws + E;
    unsigned short* Vws = Kws + E;
    unsigned short* AOw = Vws + E;

    qkv_gemm_bf16<<<dim3(512, 12), 128, 0, stream>>>(x, qkv_w, Qws, Kws, Vws);
    attn_flash  <<<dim3(32, 32),  128, 0, stream>>>(Qws, Kws, Vws, AOw);
    proj_gemm_bf16<<<dim3(512, 4), 128, 0, stream>>>(AOw, proj_w, proj_b, out);
}
